// ADRHeteroGCN_1468878815453
// MI455X (gfx1250) — matrix operand order repaired and checked by reading, not compiled, not run on hardware
//
#include <hip/hip_runtime.h>
#include <hip/hip_bf16.h>

typedef __bf16 bf16;
typedef __attribute__((ext_vector_type(16))) __bf16 v16bf;
typedef __attribute__((ext_vector_type(8)))  __bf16 v8bf;
typedef __attribute__((ext_vector_type(8)))  float  v8f;

#define D_IN 128
#define HDIM 256

__device__ __forceinline__ void atomicAddF(float* p, float v) {
    unsafeAtomicAdd(p, v);   // lowers to global_atomic_add_f32 on gfx1250
}

// ---------------------------------------------------------------------------
// fill zero
// ---------------------------------------------------------------------------
__global__ void fill_zero(float* __restrict__ p, long long n) {
    long long t = (long long)blockIdx.x * blockDim.x + threadIdx.x;
    if (t < n) p[t] = 0.0f;
}

// ---------------------------------------------------------------------------
// Convert+transpose per-relation Wl [R,K,H] fp32 -> [R,H,K] bf16
// ---------------------------------------------------------------------------
__global__ void prep_wl(const float* __restrict__ W, bf16* __restrict__ Wt,
                        int R, int K, int H) {
    long long t = (long long)blockIdx.x * blockDim.x + threadIdx.x;
    long long tot = (long long)R * K * H;
    if (t >= tot) return;
    int i = (int)(t / ((long long)K * H));
    int rem = (int)(t % ((long long)K * H));
    int k = rem / H, n = rem % H;
    Wt[(size_t)i * H * K + (size_t)n * K + k] = (bf16)W[t];
}

// ---------------------------------------------------------------------------
// Per-dst-type summed root weights (transposed bf16) + summed bias.
// dst types: drug=0 (rels 4,5), protein=1 (rels 1,6), side_effect=2 (rels 0,3),
// pathway=3 (rel 2).
// ---------------------------------------------------------------------------
__global__ void prep_wr(const float* __restrict__ Wr, const float* __restrict__ b,
                        bf16* __restrict__ Wt, float* __restrict__ bsum,
                        int K, int H) {
    const int nrel[4]    = {2, 2, 2, 1};
    const int rels[4][2] = {{4, 5}, {1, 6}, {0, 3}, {2, 0}};
    long long t = (long long)blockIdx.x * blockDim.x + threadIdx.x;
    long long tot = 4ll * K * H;
    if (t >= tot) return;
    int ty = (int)(t / ((long long)K * H));
    int rem = (int)(t % ((long long)K * H));
    int k = rem / H, n = rem % H;
    float s = 0.0f, bs = 0.0f;
    for (int j = 0; j < nrel[ty]; ++j) {
        int i = rels[ty][j];
        s  += Wr[(size_t)i * K * H + (size_t)k * H + n];
        bs += b[i * H + n];
    }
    Wt[(size_t)ty * H * K + (size_t)n * K + k] = (bf16)s;
    if (k == 0) bsum[ty * H + n] = bs;
}

// ---------------------------------------------------------------------------
// Edge scatter-add: one thread per (edge, feature). Hardware fp32 atomics.
// ---------------------------------------------------------------------------
__global__ void scatter_add(const float* __restrict__ X,
                            const int* __restrict__ src, const int* __restrict__ dst,
                            long long E, int D,
                            float* __restrict__ agg, float* __restrict__ cnt) {
    long long t = (long long)blockIdx.x * blockDim.x + threadIdx.x;
    long long tot = E * D;
    if (t >= tot) return;
    long long e = t / D;
    int d = (int)(t % D);
    int s  = src[e];
    int dd = dst[e];
    atomicAddF(&agg[(size_t)dd * D + d], X[(size_t)s * D + d]);
    if (d == 0) atomicAddF(&cnt[dd], 1.0f);
}

__global__ void invert_cnt(float* __restrict__ cnt, int n) {
    int t = blockIdx.x * blockDim.x + threadIdx.x;
    if (t < n) cnt[t] = 1.0f / fmaxf(cnt[t], 1.0f);
}

// ---------------------------------------------------------------------------
// WMMA bf16 GEMM: C[N,NC] (+)= (A[N,K] * rowscale) @ Bt[NC,K]^T (+ bias)
// Block = 256 threads = 8 waves; block tile = 16 rows x 128 cols.
// Each wave: one 16x16 tile via v_wmma_f32_16x16x32_bf16, K-loop step 32.
// ---------------------------------------------------------------------------
__global__ void gemm_wmma(const float* __restrict__ A,
                          const float* __restrict__ rowscale,   // inv counts or null
                          const bf16* __restrict__ Bt,          // [NC][K] bf16
                          const float* __restrict__ bias,       // [NC] or null
                          float* __restrict__ C,
                          int Nrows, int K, int NC, int accumulate) {
    __shared__ bf16 As[16 * 256];                 // 16 rows x K (K<=256), 8 KB
    const int row0    = blockIdx.x * 16;
    const int colBase = blockIdx.y * 128;
    const int tid = threadIdx.x;

    // Stage A tile: fp32 -> bf16, fused mean scaling, zero-pad OOB rows.
    const int total = 16 * K;
    for (int idx = tid; idx < total; idx += 256) {
        int r = idx / K, k = idx % K;
        int gr = row0 + r;
        float v = 0.0f;
        if (gr < Nrows) {
            v = A[(size_t)gr * K + k];
            if (rowscale) v *= rowscale[gr];
        }
        As[r * K + k] = (bf16)v;
    }
    __syncthreads();

    const int wave = tid >> 5;
    const int lane = tid & 31;
    const int m    = lane & 15;
    const int half = lane >> 4;
    const int col0 = colBase + wave * 16;
    const int col  = col0 + m;

    v8f c = {};
    for (int kk = 0; kk < K; kk += 32) {
        // A fragment: row m; K = kk+half*8+{0..7} and kk+16+half*8+{0..7}
        v8bf alo = *(const v8bf*)(&As[m * K + kk + half * 8]);
        v8bf ahi = *(const v8bf*)(&As[m * K + kk + 16 + half * 8]);
        v16bf a;
        for (int j = 0; j < 8; ++j) { a[j] = alo[j]; a[8 + j] = ahi[j]; }
        // B fragment: column 'col'; K = kk+half*8+{0..7} and kk+16+half*8+{0..7}
        const bf16* bp = &Bt[(size_t)col * K + kk + half * 8];
        v8bf blo = *(const v8bf*)(bp);
        v8bf bhi = *(const v8bf*)(bp + 16);
        v16bf b;
        for (int j = 0; j < 8; ++j) { b[j] = blo[j]; b[8 + j] = bhi[j]; }
        c = __builtin_amdgcn_wmma_f32_16x16x32_bf16(
                false, a, false, b, (short)0, c, false, false);
    }

    // Epilogue: D layout row = vgpr + 8*half, col = lane%16
    for (int vv = 0; vv < 8; ++vv) {
        int r = row0 + vv + 8 * half;
        if (r < Nrows) {
            size_t off = (size_t)r * NC + col;
            float val = c[vv];
            if (accumulate)   val += C[off];
            else if (bias)    val += bias[col];
            C[off] = val;
        }
    }
}

// ---------------------------------------------------------------------------
// BatchNorm (training, biased var) + ReLU.
// ---------------------------------------------------------------------------
__global__ void bn_stats(const float* __restrict__ X, int N, int D,
                         float* __restrict__ stats) {
    int col = threadIdx.x;                 // blockDim.x == D == 256
    float s = 0.0f, s2 = 0.0f;
    for (int r = blockIdx.x; r < N; r += gridDim.x) {
        float v = X[(size_t)r * D + col];
        s += v; s2 += v * v;
    }
    atomicAddF(&stats[col], s);
    atomicAddF(&stats[D + col], s2);
}

__global__ void bn_apply(float* __restrict__ X, const float* __restrict__ stats,
                         const float* __restrict__ g, const float* __restrict__ beta,
                         int N, int D) {
    long long t = (long long)blockIdx.x * blockDim.x + threadIdx.x;
    long long tot = (long long)N * D;
    if (t >= tot) return;
    int col = (int)(t % D);
    float invN = 1.0f / (float)N;
    float mu  = stats[col] * invN;
    float var = stats[D + col] * invN - mu * mu;
    float v = (X[t] - mu) * rsqrtf(var + 1e-5f) * g[col] + beta[col];
    X[t] = fmaxf(v, 0.0f);
}

// ---------------------------------------------------------------------------
// Host orchestration
// ---------------------------------------------------------------------------
static inline unsigned blocks_for(long long n, int bs = 256) {
    return (unsigned)((n + bs - 1) / bs);
}

extern "C" void kernel_launch(void* const* d_in, const int* in_sizes, int n_in,
                              void* d_out, int out_size, void* d_ws, size_t ws_size,
                              hipStream_t stream) {
    const int NT[4]     = {20000, 50000, 10000, 3000};
    const int relsrc[7] = {0, 0, 1, 1, 2, 1, 3};
    const int reldst[7] = {2, 1, 3, 2, 0, 0, 1};

    const float* X1[4];
    for (int t = 0; t < 4; ++t) X1[t] = (const float*)d_in[t];
    const int* SRC[7]; const int* DST[7]; long long EC[7];
    for (int i = 0; i < 7; ++i) {
        SRC[i] = (const int*)d_in[4 + 2 * i];
        DST[i] = (const int*)d_in[5 + 2 * i];
        EC[i]  = in_sizes[4 + 2 * i];
    }
    const float* W1l = (const float*)d_in[18];
    const float* W1r = (const float*)d_in[19];
    const float* b1  = (const float*)d_in[20];
    const float* W2l = (const float*)d_in[21];
    const float* W2r = (const float*)d_in[22];
    const float* b2  = (const float*)d_in[23];
    const float* g1  = (const float*)d_in[24];
    const float* be1 = (const float*)d_in[25];
    const float* g2  = (const float*)d_in[26];
    const float* be2 = (const float*)d_in[27];

    // Workspace carve-up (256B aligned)
    char* w = (char*)d_ws;
    auto carve = [&](size_t bytes) -> void* {
        void* p = (void*)w;
        w += (bytes + 255) & ~(size_t)255;
        return p;
    };
    bf16*  Wl1t  = (bf16*) carve((size_t)7 * D_IN * HDIM * sizeof(bf16));
    bf16*  Wr1t  = (bf16*) carve((size_t)4 * D_IN * HDIM * sizeof(bf16));
    bf16*  Wl2t  = (bf16*) carve((size_t)7 * HDIM * HDIM * sizeof(bf16));
    bf16*  Wr2t  = (bf16*) carve((size_t)4 * HDIM * HDIM * sizeof(bf16));
    float* bsum1 = (float*)carve((size_t)4 * HDIM * sizeof(float));
    float* bsum2 = (float*)carve((size_t)4 * HDIM * sizeof(float));
    float* stats = (float*)carve((size_t)2 * HDIM * sizeof(float));
    float* cnt   = (float*)carve((size_t)50000 * sizeof(float));
    float* agg   = (float*)carve((size_t)50000 * HDIM * sizeof(float));
    float* h1    = (float*)carve((size_t)83000 * HDIM * sizeof(float));

    // Per-type row offsets (h1 and d_out share layout: [Nt,256] concatenated)
    size_t off[4] = {0, (size_t)20000 * HDIM, (size_t)70000 * HDIM, (size_t)80000 * HDIM};
    float* OUT = (float*)d_out;

    // ---- Weight prep ----
    prep_wl<<<blocks_for(7ll * D_IN * HDIM), 256, 0, stream>>>(W1l, Wl1t, 7, D_IN, HDIM);
    prep_wl<<<blocks_for(7ll * HDIM * HDIM), 256, 0, stream>>>(W2l, Wl2t, 7, HDIM, HDIM);
    prep_wr<<<blocks_for(4ll * D_IN * HDIM), 256, 0, stream>>>(W1r, b1, Wr1t, bsum1, D_IN, HDIM);
    prep_wr<<<blocks_for(4ll * HDIM * HDIM), 256, 0, stream>>>(W2r, b2, Wr2t, bsum2, HDIM, HDIM);

    // ======================= Layer 1 (K = 128) ===========================
    for (int t = 0; t < 4; ++t) {   // root: h1_t = x_t @ WrSum_t + bsum_t
        dim3 grid(blocks_for(NT[t], 16), HDIM / 128);
        gemm_wmma<<<grid, 256, 0, stream>>>(X1[t], nullptr,
            Wr1t + (size_t)t * HDIM * D_IN, bsum1 + t * HDIM,
            h1 + off[t], NT[t], D_IN, HDIM, 0);
    }
    for (int i = 0; i < 7; ++i) {   // aggregation terms
        int dt = reldst[i], st = relsrc[i];
        long long na = (long long)NT[dt] * D_IN;
        fill_zero<<<blocks_for(na), 256, 0, stream>>>(agg, na);
        fill_zero<<<blocks_for(NT[dt]), 256, 0, stream>>>(cnt, NT[dt]);
        scatter_add<<<blocks_for(EC[i] * D_IN), 256, 0, stream>>>(
            X1[st], SRC[i], DST[i], EC[i], D_IN, agg, cnt);
        invert_cnt<<<blocks_for(NT[dt]), 256, 0, stream>>>(cnt, NT[dt]);
        dim3 grid(blocks_for(NT[dt], 16), HDIM / 128);
        gemm_wmma<<<grid, 256, 0, stream>>>(agg, cnt,
            Wl1t + (size_t)i * HDIM * D_IN, nullptr,
            h1 + off[dt], NT[dt], D_IN, HDIM, 1);
    }
    for (int t = 0; t < 4; ++t) {   // BN + ReLU in place
        fill_zero<<<blocks_for(2 * HDIM), 256, 0, stream>>>(stats, 2 * HDIM);
        bn_stats<<<256, HDIM, 0, stream>>>(h1 + off[t], NT[t], HDIM, stats);
        bn_apply<<<blocks_for((long long)NT[t] * HDIM), 256, 0, stream>>>(
            h1 + off[t], stats, g1, be1, NT[t], HDIM);
    }

    // ======================= Layer 2 (K = 256) ===========================
    for (int t = 0; t < 4; ++t) {
        dim3 grid(blocks_for(NT[t], 16), HDIM / 128);
        gemm_wmma<<<grid, 256, 0, stream>>>(h1 + off[t], nullptr,
            Wr2t + (size_t)t * HDIM * HDIM, bsum2 + t * HDIM,
            OUT + off[t], NT[t], HDIM, HDIM, 0);
    }
    for (int i = 0; i < 7; ++i) {
        int dt = reldst[i], st = relsrc[i];
        long long na = (long long)NT[dt] * HDIM;
        fill_zero<<<blocks_for(na), 256, 0, stream>>>(agg, na);
        fill_zero<<<blocks_for(NT[dt]), 256, 0, stream>>>(cnt, NT[dt]);
        scatter_add<<<blocks_for(EC[i] * HDIM), 256, 0, stream>>>(
            h1 + off[st], SRC[i], DST[i], EC[i], HDIM, agg, cnt);
        invert_cnt<<<blocks_for(NT[dt]), 256, 0, stream>>>(cnt, NT[dt]);
        dim3 grid(blocks_for(NT[dt], 16), HDIM / 128);
        gemm_wmma<<<grid, 256, 0, stream>>>(agg, cnt,
            Wl2t + (size_t)i * HDIM * HDIM, nullptr,
            OUT + off[dt], NT[dt], HDIM, HDIM, 1);
    }
    for (int t = 0; t < 4; ++t) {
        fill_zero<<<blocks_for(2 * HDIM), 256, 0, stream>>>(stats, 2 * HDIM);
        bn_stats<<<256, HDIM, 0, stream>>>(OUT + off[t], NT[t], HDIM, stats);
        bn_apply<<<blocks_for((long long)NT[t] * HDIM), 256, 0, stream>>>(
            OUT + off[t], stats, g2, be2, NT[t], HDIM);
    }
}
